// SelfAttention_17738214932656
// MI455X (gfx1250) — matrix operand order repaired and checked by reading, not compiled, not run on hardware
//
#include <hip/hip_runtime.h>

#define N_TOK  2048
#define C_DIM  768
#define CS_DIM 384
#define H_HEADS 16
#define D_HEAD 48

typedef __bf16 bf16_t;
typedef __attribute__((ext_vector_type(16))) __bf16 v16bf;
typedef __attribute__((ext_vector_type(8)))  __bf16 v8bf;
typedef __attribute__((ext_vector_type(4)))  __bf16 v4bf;
typedef __attribute__((ext_vector_type(8)))  float  v8f;

static __device__ __forceinline__ float sigmoidf_(float x) {
    return 1.0f / (1.0f + __expf(-x));
}

static __device__ __forceinline__ v16bf combine16(v8bf lo, v8bf hi) {
    v16bf r;
#pragma unroll
    for (int i = 0; i < 8; ++i) { r[i] = lo[i]; r[8 + i] = hi[i]; }
    return r;
}

static __device__ __forceinline__ v8f wmma_bf16(v16bf a, v16bf b, v8f c) {
    // v_wmma_f32_16x16x32_bf16  D = A(16x32) * B(32x16) + C(16x16)
    return __builtin_amdgcn_wmma_f32_16x16x32_bf16(
        /*neg_a=*/false, a, /*neg_b=*/false, b,
        /*c_mod=*/(short)0, c, /*reuse_a=*/false, /*reuse_b=*/false);
}

// A-operand fragment (16x32 bf16), row-major A with leading dim lda.
// Lane m = lane&15; elements 0..7 -> K = (lane<16?0:8)+i ; 8..15 -> K = (lane<16?16:24)+i
static __device__ __forceinline__ v16bf load_a_frag(const bf16_t* A, int lda,
                                                    int row0, int k0, int lane) {
    int m  = lane & 15;
    int b0 = (lane < 16) ? 0 : 8;
    const bf16_t* p = A + (size_t)(row0 + m) * lda + k0;
    v8bf lo = *(const v8bf*)(p + b0);
    v8bf hi = *(const v8bf*)(p + b0 + 16);
    return combine16(lo, hi);
}

// B-operand fragment (32x16 bf16) for computing A @ W^T where W is (Ncols x K) row-major:
// B(k,n) = W[n][k].  Lane n = lane&15; elements 0..7 -> K = (lane<16?0:8)+i ; 8..15 -> K = (lane<16?16:24)+i
static __device__ __forceinline__ v16bf load_bT_frag(const bf16_t* W, int ldw,
                                                     int col0, int k0, int lane) {
    int n  = col0 + (lane & 15);
    int kk = k0 + ((lane < 16) ? 0 : 8);
    const bf16_t* p = W + (size_t)n * ldw + kk;
    v8bf lo = *(const v8bf*)p;
    v8bf hi = *(const v8bf*)(p + 16);
    return combine16(lo, hi);
}

// ---------------------------------------------------------------- conversions
__global__ void cvt_f32_bf16_kernel(const float* __restrict__ s,
                                    bf16_t* __restrict__ d, int n) {
    for (int i = blockIdx.x * blockDim.x + threadIdx.x; i < n;
         i += gridDim.x * blockDim.x)
        d[i] = (__bf16)s[i];
}

// ---------------------------------------------------------------- layer norms
__global__ __launch_bounds__(256) void ln_x_kernel(const float* __restrict__ x,
                                                   float* __restrict__ xn) {
    __shared__ float red[256];
    int row = blockIdx.x, tid = threadIdx.x;
    const float* xr = x + (size_t)row * C_DIM;
    float v[3];
#pragma unroll
    for (int i = 0; i < 3; ++i) v[i] = xr[tid + i * 256];
    red[tid] = v[0] + v[1] + v[2];
    __syncthreads();
    for (int o = 128; o > 0; o >>= 1) {
        if (tid < o) red[tid] += red[tid + o];
        __syncthreads();
    }
    float mean = red[0] * (1.0f / C_DIM);
    __syncthreads();
    float q = 0.f;
#pragma unroll
    for (int i = 0; i < 3; ++i) q += (v[i] - mean) * (v[i] - mean);
    red[tid] = q;
    __syncthreads();
    for (int o = 128; o > 0; o >>= 1) {
        if (tid < o) red[tid] += red[tid + o];
        __syncthreads();
    }
    float rstd = rsqrtf(red[0] * (1.0f / C_DIM) + 1e-5f);
    float* out = xn + (size_t)row * C_DIM;
#pragma unroll
    for (int i = 0; i < 3; ++i) out[tid + i * 256] = (v[i] - mean) * rstd;
}

__global__ __launch_bounds__(128) void ln_sc_kernel(const float* __restrict__ sc,
                                                    const float* __restrict__ w,
                                                    bf16_t* __restrict__ scn) {
    __shared__ float red[128];
    int row = blockIdx.x, tid = threadIdx.x;
    const float* xr = sc + (size_t)row * CS_DIM;
    float v[3];
#pragma unroll
    for (int i = 0; i < 3; ++i) v[i] = xr[tid + i * 128];
    red[tid] = v[0] + v[1] + v[2];
    __syncthreads();
    for (int o = 64; o > 0; o >>= 1) {
        if (tid < o) red[tid] += red[tid + o];
        __syncthreads();
    }
    float mean = red[0] * (1.0f / CS_DIM);
    __syncthreads();
    float q = 0.f;
#pragma unroll
    for (int i = 0; i < 3; ++i) q += (v[i] - mean) * (v[i] - mean);
    red[tid] = q;
    __syncthreads();
    for (int o = 64; o > 0; o >>= 1) {
        if (tid < o) red[tid] += red[tid + o];
        __syncthreads();
    }
    float rstd = rsqrtf(red[0] * (1.0f / CS_DIM) + 1e-5f);
    bf16_t* out = scn + (size_t)row * CS_DIM;
#pragma unroll
    for (int i = 0; i < 3; ++i) {
        int c = tid + i * 128;
        out[c] = (__bf16)((v[i] - mean) * rstd * w[c]);
    }
}

// -------------------------------------------- fused AdaLN (dual WMMA GEMM)
// xm = sigmoid(scn@scale_w^T + scale_b) * xn + scn@bias_w^T
__global__ __launch_bounds__(256) void adaln_kernel(
    const bf16_t* __restrict__ scn, const bf16_t* __restrict__ Wsc,
    const bf16_t* __restrict__ Wbi, const float* __restrict__ scale_b,
    const float* __restrict__ xn, bf16_t* __restrict__ xm) {
    int lane = threadIdx.x & 31, wave = threadIdx.x >> 5;
    int tiles_n = C_DIM >> 4;
    int tile = blockIdx.x * 8 + wave;
    int tm = tile / tiles_n, tn = tile - tm * tiles_n;
    int row0 = tm * 16, col0 = tn * 16;
    v8f aS = {}, aB = {};
    for (int k0 = 0; k0 < CS_DIM; k0 += 32) {
        v16bf a = load_a_frag(scn, CS_DIM, row0, k0, lane);
        aS = wmma_bf16(a, load_bT_frag(Wsc, CS_DIM, col0, k0, lane), aS);
        aB = wmma_bf16(a, load_bT_frag(Wbi, CS_DIM, col0, k0, lane), aB);
    }
    int n = col0 + (lane & 15);
    int mbase = (lane < 16) ? 0 : 8;
    float sb = scale_b[n];
#pragma unroll
    for (int r = 0; r < 8; ++r) {
        size_t idx = (size_t)(row0 + mbase + r) * C_DIM + n;
        xm[idx] = (__bf16)(sigmoidf_(aS[r] + sb) * xn[idx] + aB[r]);
    }
}

// -------------------------------------------- fused q/k/v/gate projections
// One A-fragment (xm tile) feeds 4 WMMAs per k-step: 4x WMMA density per byte.
__global__ __launch_bounds__(256) void qkvg_kernel(
    const bf16_t* __restrict__ xm, const bf16_t* __restrict__ Wq,
    const bf16_t* __restrict__ Wk, const bf16_t* __restrict__ Wv,
    const bf16_t* __restrict__ Wg, const float* __restrict__ q_b,
    bf16_t* __restrict__ qo, bf16_t* __restrict__ ko, bf16_t* __restrict__ vo,
    float* __restrict__ gate, float qscale) {
    int lane = threadIdx.x & 31, wave = threadIdx.x >> 5;
    int tiles_n = C_DIM >> 4;
    int tile = blockIdx.x * 8 + wave;
    int tm = tile / tiles_n, tn = tile - tm * tiles_n;
    int row0 = tm * 16, col0 = tn * 16;
    v8f aq = {}, ak = {}, av = {}, ag = {};
    for (int k0 = 0; k0 < C_DIM; k0 += 32) {
        v16bf a = load_a_frag(xm, C_DIM, row0, k0, lane);
        aq = wmma_bf16(a, load_bT_frag(Wq, C_DIM, col0, k0, lane), aq);
        ak = wmma_bf16(a, load_bT_frag(Wk, C_DIM, col0, k0, lane), ak);
        av = wmma_bf16(a, load_bT_frag(Wv, C_DIM, col0, k0, lane), av);
        ag = wmma_bf16(a, load_bT_frag(Wg, C_DIM, col0, k0, lane), ag);
    }
    int n = col0 + (lane & 15);
    int mbase = (lane < 16) ? 0 : 8;
    float bq = q_b[n];
#pragma unroll
    for (int r = 0; r < 8; ++r) {
        size_t idx = (size_t)(row0 + mbase + r) * C_DIM + n;
        qo[idx]   = (__bf16)((aq[r] + bq) * qscale);
        ko[idx]   = (__bf16)ak[r];
        vo[idx]   = (__bf16)av[r];
        gate[idx] = sigmoidf_(ag[r]);
    }
}

// -------------------------------------------- generic bf16 WMMA GEMM: A @ W^T
// MODE 1: out f32 = sigmoid(acc + bias)
// MODE 2: out f32 = acc * mul[row][col]
template <int MODE>
__global__ __launch_bounds__(256) void gemm_kernel(
    const bf16_t* __restrict__ A, int lda, const bf16_t* __restrict__ W, int ldw,
    const float* __restrict__ bias, const float* __restrict__ mul,
    float* __restrict__ out, int Ncols, int K) {
    int lane = threadIdx.x & 31, wave = threadIdx.x >> 5;
    int tiles_n = Ncols >> 4;
    int tile = blockIdx.x * 8 + wave;
    int tm = tile / tiles_n, tn = tile - tm * tiles_n;
    int row0 = tm * 16, col0 = tn * 16;
    v8f acc = {};
    for (int k0 = 0; k0 < K; k0 += 32) {
        v16bf a = load_a_frag(A, lda, row0, k0, lane);
        v16bf b = load_bT_frag(W, ldw, col0, k0, lane);
        acc = wmma_bf16(a, b, acc);
    }
    int n = col0 + (lane & 15);
    int mbase = (lane < 16) ? 0 : 8;
    float bi = bias ? bias[n] : 0.0f;
#pragma unroll
    for (int r = 0; r < 8; ++r) {
        size_t idx = (size_t)(row0 + mbase + r) * Ncols + n;
        if (MODE == 1)
            out[idx] = sigmoidf_(acc[r] + bi);
        else
            out[idx] = acc[r] * mul[idx];
    }
}

// -------------------------------------------- flash attention, one-pass
// grid = (N/64, H), block = 128 (4 waves, one 16-row q tile per wave).
// K tile staged with gfx1250 async-to-LDS copies (ASYNCcnt); V staged with a
// VGPR transpose; pair_logits streamed once; gate folded into the epilogue.
__global__ __launch_bounds__(128) void attn_kernel(
    const bf16_t* __restrict__ Q, const bf16_t* __restrict__ Km,
    const bf16_t* __restrict__ Vm, const float* __restrict__ pair,
    const float* __restrict__ mask, const float* __restrict__ gate,
    bf16_t* __restrict__ Ob) {
    __shared__ bf16_t Ks[32][64];      // [key][d], d padded 48->64 with zeros
    __shared__ bf16_t Vt[64][32];      // [d][key] transposed, rows 48..63 zero
    __shared__ bf16_t Ps[4][16][32];   // per-wave P re-layout staging

    int tid = threadIdx.x, lane = tid & 31, wave = tid >> 5;
    int h = blockIdx.y;
    int q0 = blockIdx.x * 64 + wave * 16;
    int m = lane & 15;
    int half = (lane < 16) ? 0 : 1;

    // one-time zero fill of the d=48..63 pad regions (never rewritten)
    {
        unsigned long long z = 0ull;
        int key = tid >> 2, dp = 48 + 4 * (tid & 3);
        *(unsigned long long*)&Ks[key][dp] = z;
        int rowp = 48 + (tid >> 3), cp = 4 * (tid & 7);
        *(unsigned long long*)&Vt[rowp][cp] = z;
    }

    // Q fragments for d-chunks [0,32) and [32,64) (zero-padded past 48)
    v16bf qf[2];
#pragma unroll
    for (int dc = 0; dc < 2; ++dc) {
        const bf16_t* qrow = Q + (size_t)(q0 + m) * C_DIM + h * D_HEAD;
        int b0 = half ? 8 : 0;
        v16bf f;
#pragma unroll
        for (int i = 0; i < 8; ++i) {
            int d0 = dc * 32 + b0 + i;
            int d1 = dc * 32 + b0 + 16 + i;
            f[i]     = (d0 < D_HEAD) ? qrow[d0] : (__bf16)0.0f;
            f[8 + i] = (d1 < D_HEAD) ? qrow[d1] : (__bf16)0.0f;
        }
        qf[dc] = f;
    }

    float rmax[8], rsum[8];
    v8f o0 = {}, o1 = {}, o2 = {};
#pragma unroll
    for (int r = 0; r < 8; ++r) { rmax[r] = -1e30f; rsum[r] = 0.0f; }

    for (int kb = 0; kb < N_TOK; kb += 32) {
        __syncthreads();  // previous-iteration LDS reads done; pad fill done

        // K tile: 32 keys x 96B = 192 16B chunks, async global->LDS (no VGPRs)
        for (int c = tid; c < 192; c += 128) {
            int key = c / 6, part = c % 6;
            const bf16_t* g = Km + (size_t)(kb + key) * C_DIM + h * D_HEAD + part * 8;
            unsigned ldsoff = (unsigned)(size_t)&Ks[key][part * 8];
            asm volatile("global_load_async_to_lds_b128 %0, %1, off"
                         :: "v"(ldsoff), "v"((unsigned long long)(size_t)g)
                         : "memory");
        }
        // V tile: per-thread 12 elems (3x 8B loads) + transposed b16 stores
        {
            int key = tid >> 2, seg = (tid & 3) * 12;
            const bf16_t* g = Vm + (size_t)(kb + key) * C_DIM + h * D_HEAD + seg;
            v4bf a0 = *(const v4bf*)g;
            v4bf a1 = *(const v4bf*)(g + 4);
            v4bf a2 = *(const v4bf*)(g + 8);
#pragma unroll
            for (int j = 0; j < 4; ++j) {
                Vt[seg + j][key]     = a0[j];
                Vt[seg + 4 + j][key] = a1[j];
                Vt[seg + 8 + j][key] = a2[j];
            }
        }
        asm volatile("s_wait_asynccnt 0x0" ::: "memory");
        __syncthreads();

        float mb0 = 1e9f * (mask[kb + m] - 1.0f);
        float mb1 = 1e9f * (mask[kb + 16 + m] - 1.0f);

        // S = Q K^T over padded d=64 (2 chunks), keys split 0..15 / 16..31
        v8f s0 = {}, s1 = {};
#pragma unroll
        for (int dc = 0; dc < 2; ++dc) {
            int kd = dc * 32 + (half ? 8 : 0);
            v16bf b0f = combine16(*(const v8bf*)&Ks[m][kd],
                                  *(const v8bf*)&Ks[m][kd + 16]);
            v16bf b1f = combine16(*(const v8bf*)&Ks[m + 16][kd],
                                  *(const v8bf*)&Ks[m + 16][kd + 16]);
            s0 = wmma_bf16(qf[dc], b0f, s0);
            s1 = wmma_bf16(qf[dc], b1f, s1);
        }

        // pair_logits + mask bias, online softmax (row = r + 8*half)
#pragma unroll
        for (int r = 0; r < 8; ++r) {
            int row = q0 + half * 8 + r;
            const float* pp = pair + ((size_t)h * N_TOK + row) * N_TOK + kb;
            float x0 = s0[r] + pp[m] + mb0;
            float x1 = s1[r] + pp[16 + m] + mb1;
            float mx = fmaxf(x0, x1);
#pragma unroll
            for (int off = 1; off < 16; off <<= 1)
                mx = fmaxf(mx, __shfl_xor(mx, off, 32));
            float mnew = fmaxf(rmax[r], mx);
            float scale = __expf(rmax[r] - mnew);
            rmax[r] = mnew;
            float p0 = __expf(x0 - mnew), p1 = __expf(x1 - mnew);
            float ps = p0 + p1;
#pragma unroll
            for (int off = 1; off < 16; off <<= 1)
                ps += __shfl_xor(ps, off, 32);
            rsum[r] = rsum[r] * scale + ps;
            o0[r] *= scale; o1[r] *= scale; o2[r] *= scale;
            Ps[wave][half * 8 + r][m]      = (__bf16)p0;
            Ps[wave][half * 8 + r][16 + m] = (__bf16)p1;
        }

        // P (16x32, A layout) @ V (32x48) -> accumulate 3 d-chunks
        {
            int b0 = half ? 8 : 0;
            v16bf pf = combine16(*(const v8bf*)&Ps[wave][m][b0],
                                 *(const v8bf*)&Ps[wave][m][b0 + 16]);
            int kk = half ? 8 : 0;
#pragma unroll
            for (int dc = 0; dc < 3; ++dc) {
                int dd = dc * 16 + m;
                v16bf bf_ = combine16(*(const v8bf*)&Vt[dd][kk],
                                      *(const v8bf*)&Vt[dd][kk + 16]);
                if (dc == 0)      o0 = wmma_bf16(pf, bf_, o0);
                else if (dc == 1) o1 = wmma_bf16(pf, bf_, o1);
                else              o2 = wmma_bf16(pf, bf_, o2);
            }
        }
    }

    // epilogue: normalize + sigmoid-gate multiply, write bf16 for t2 GEMM
#pragma unroll
    for (int r = 0; r < 8; ++r) {
        int row = q0 + half * 8 + r;
        float inv = 1.0f / rsum[r];
        size_t base = (size_t)row * C_DIM + h * D_HEAD + m;
        Ob[base]      = (__bf16)(o0[r] * inv * gate[base]);
        Ob[base + 16] = (__bf16)(o1[r] * inv * gate[base + 16]);
        Ob[base + 32] = (__bf16)(o2[r] * inv * gate[base + 32]);
    }
}

// ============================================================= host launcher
extern "C" void kernel_launch(void* const* d_in, const int* in_sizes, int n_in,
                              void* d_out, int out_size, void* d_ws,
                              size_t ws_size, hipStream_t stream) {
    (void)in_sizes; (void)n_in; (void)out_size; (void)ws_size;
    const float* x        = (const float*)d_in[0];
    const float* mask     = (const float*)d_in[1];
    const float* pair     = (const float*)d_in[2];
    const float* sc       = (const float*)d_in[3];
    const float* q_w      = (const float*)d_in[4];
    const float* q_b      = (const float*)d_in[5];
    const float* k_w      = (const float*)d_in[6];
    const float* v_w      = (const float*)d_in[7];
    const float* gate_w   = (const float*)d_in[8];
    const float* sc_ln_w  = (const float*)d_in[9];
    const float* scale_w  = (const float*)d_in[10];
    const float* scale_b  = (const float*)d_in[11];
    const float* bias_w   = (const float*)d_in[12];
    const float* t2_w     = (const float*)d_in[13];
    const float* az_w     = (const float*)d_in[14];
    const float* az_b     = (const float*)d_in[15];

    char* ws = (char*)d_ws;
    size_t off = 0;
    auto take = [&](size_t bytes) {
        off = (off + 255) & ~(size_t)255;
        size_t o = off; off += bytes; return o;
    };
    const size_t CC = (size_t)C_DIM * C_DIM, CCS = (size_t)C_DIM * CS_DIM;
    const size_t NC = (size_t)N_TOK * C_DIM, NCS = (size_t)N_TOK * CS_DIM;

    bf16_t* w_q     = (bf16_t*)(ws + take(CC * 2));
    bf16_t* w_k     = (bf16_t*)(ws + take(CC * 2));
    bf16_t* w_v     = (bf16_t*)(ws + take(CC * 2));
    bf16_t* w_gate  = (bf16_t*)(ws + take(CC * 2));
    bf16_t* w_t2    = (bf16_t*)(ws + take(CC * 2));
    bf16_t* w_scale = (bf16_t*)(ws + take(CCS * 2));
    bf16_t* w_bias  = (bf16_t*)(ws + take(CCS * 2));
    bf16_t* w_az    = (bf16_t*)(ws + take(CCS * 2));
    bf16_t* sc_raw  = (bf16_t*)(ws + take(NCS * 2));
    bf16_t* scn     = (bf16_t*)(ws + take(NCS * 2));
    float*  xn      = (float*) (ws + take(NC * 4));
    bf16_t* xm      = (bf16_t*)(ws + take(NC * 2));
    bf16_t* qb      = (bf16_t*)(ws + take(NC * 2));
    bf16_t* kb      = (bf16_t*)(ws + take(NC * 2));
    bf16_t* vb      = (bf16_t*)(ws + take(NC * 2));
    float*  gate    = (float*) (ws + take(NC * 4));
    float*  adaz    = (float*) (ws + take(NC * 4));
    bf16_t* o_bf    = (bf16_t*)(ws + take(NC * 2));

    auto cvt = [&](const float* s, bf16_t* d, size_t n) {
        cvt_f32_bf16_kernel<<<(int)((n + 1023) / 1024), 256, 0, stream>>>(s, d, (int)n);
    };
    cvt(q_w, w_q, CC);       cvt(k_w, w_k, CC);     cvt(v_w, w_v, CC);
    cvt(gate_w, w_gate, CC); cvt(t2_w, w_t2, CC);
    cvt(scale_w, w_scale, CCS); cvt(bias_w, w_bias, CCS); cvt(az_w, w_az, CCS);
    cvt(sc, sc_raw, NCS);

    ln_x_kernel<<<N_TOK, 256, 0, stream>>>(x, xn);
    ln_sc_kernel<<<N_TOK, 128, 0, stream>>>(sc, sc_ln_w, scn);

    const int gemm_blocks = (N_TOK / 16) * (C_DIM / 16) / 8;  // 768
    adaln_kernel<<<gemm_blocks, 256, 0, stream>>>(scn, w_scale, w_bias, scale_b, xn, xm);

    const float qscale = 0.14433756729740643f;  // 48^-0.5
    qkvg_kernel<<<gemm_blocks, 256, 0, stream>>>(xm, w_q, w_k, w_v, w_gate, q_b,
                                                 qb, kb, vb, gate, qscale);
    gemm_kernel<1><<<gemm_blocks, 256, 0, stream>>>(sc_raw, CS_DIM, w_az, CS_DIM,
                                                    az_b, nullptr, adaz, C_DIM, CS_DIM);

    dim3 agrid(N_TOK / 64, H_HEADS);
    attn_kernel<<<agrid, 128, 0, stream>>>(qb, kb, vb, pair, mask, gate, o_bf);

    gemm_kernel<2><<<gemm_blocks, 256, 0, stream>>>(o_bf, C_DIM, w_t2, C_DIM, nullptr,
                                                    adaz, (float*)d_out, C_DIM, C_DIM);
}
